// MultiHeadDotProductAttention_with_AttnWeights_88811333747460
// MI455X (gfx1250) — matrix operand order repaired and checked by reading, not compiled, not run on hardware
//
#include <hip/hip_runtime.h>

// ---------------------------------------------------------------------------
// Multi-head attention (with materialized attention weights) for gfx1250.
// - all matmuls on v_wmma_f32_16x16x32_bf16 (f32 accumulate)
// - TDM (tensor_load_to_lds) stages bf16 tiles global->LDS asynchronously
// - attention passes software-pipelined to fill WMMA->VALU hazard slots
// ---------------------------------------------------------------------------

typedef __attribute__((ext_vector_type(16))) __bf16 v16bf;
typedef __attribute__((ext_vector_type(8)))  __bf16 v8bf;
typedef __attribute__((ext_vector_type(8)))  float  v8f;
typedef __attribute__((ext_vector_type(4)))  unsigned int u32x4;
typedef __attribute__((ext_vector_type(8)))  unsigned int u32x8;

#define Bb 2
#define Ss 2048
#define Dd 1024
#define Hh 16
#define Hd 64
#define BS (Bb * Ss)      // 4096 rows
#define NK 1024           // contraction dim of both projection GEMMs
#define NN 1024           // output cols of both projection GEMMs

__device__ __forceinline__ v16bf cat16(v8bf lo, v8bf hi) {
    return __builtin_shufflevector(lo, hi, 0,1,2,3,4,5,6,7,8,9,10,11,12,13,14,15);
}

__device__ __forceinline__ v8f wmma_bf16(v16bf a, v16bf b, v8f c) {
    return __builtin_amdgcn_wmma_f32_16x16x32_bf16(
        /*neg_a=*/false, a, /*neg_b=*/false, b,
        /*c_mod=*/(short)0, c, /*reuse_a=*/false, /*reuse_b=*/false);
}

// Low 32 bits of a generic pointer to LDS == LDS byte offset (ISA aperture rule).
__device__ __forceinline__ unsigned lds_off32(const void* p) {
    return (unsigned)(unsigned long long)p;
}

// ---------------------------------------------------------------------------
// TDM: async DMA of a 2D bf16 tile (tile_d1 rows x tile_d0 elems, row stride
// stride_d0 elems) from global memory into LDS, with optional LDS row padding
// (pad after 32 DWORDs (=64 bf16) by 4 DWORDs (=8 bf16) -> LDS stride 72).
// D# packed per CDNA5 ISA 8.3/8.4.  Issue from ONE wave; EXEC is ignored by
// TDM, but the wave-level branch (s_cbranch_execz) keeps other waves away.
// ---------------------------------------------------------------------------
__device__ __forceinline__ void tdm_load_2d_bf16(
    unsigned lds_addr, const void* gaddr,
    unsigned tensor_d0, unsigned tensor_d1,
    unsigned tile_d0, unsigned tile_d1,
    unsigned stride_d0, bool pad64to72)
{
    unsigned long long ga = (unsigned long long)gaddr;
    u32x4 g0;
    g0[0] = 1u;                                            // count=1, user mode
    g0[1] = lds_addr;                                      // lds_addr[31:0]
    g0[2] = (unsigned)(ga & 0xFFFFFFFFu);                  // global_addr[31:0]
    g0[3] = (unsigned)((ga >> 32) & 0x1FFFFFFu) | (2u << 30); // ga[56:32], type=2
    u32x8 g1;
    unsigned flags = (1u << 16);                           // data_size=1 (2 bytes)
    if (pad64to72)
        flags |= (1u << 20) | (4u << 22) | (3u << 25);     // pad_en, every 32 DW, +4 DW
    g1[0] = flags;                                         // workgroup_mask=0
    g1[1] = (tensor_d0 & 0xFFFFu) << 16;                   // abar_addr=0 | td0 lo
    g1[2] = (tensor_d0 >> 16) | ((tensor_d1 & 0xFFFFu) << 16);
    g1[3] = (tensor_d1 >> 16) | (tile_d0 << 16);
    g1[4] = (tile_d1 & 0xFFFFu);                           // tile_dim2 = 0
    g1[5] = stride_d0;                                     // dim0_stride[31:0]
    g1[6] = 0u;                                            // stride hi, dim1_stride lo
    g1[7] = 0u;
    asm volatile("tensor_load_to_lds %0, %1" :: "s"(g0), "s"(g1) : "memory");
}

// ---------------------------------------------------------------------------
// Generic GEMM:  C[M x 1024] = A[M x 1024] * B[1024 x 1024], then (C+bias)*scale
// Block tile 128x64, K-tile 64, 256 threads = 8 waves, wave computes 16x64.
// IN_BF16: 0 = A fp32 in global (manual stage+convert), 1 = A bf16 (TDM stage)
// OUT_MODE: 0 = bf16 row-major; 1 = bf16 V-transposed Vt[((b*H+h)*64+e)*S+s];
//           2 = fp32 row-major (final output)
// ---------------------------------------------------------------------------
template<int IN_BF16, int OUT_MODE>
__global__ __launch_bounds__(256)
void gemm_kernel(const void* __restrict__ Xv, const float* __restrict__ W,
                 const float* __restrict__ bias, void* __restrict__ Out,
                 float scale)
{
    __shared__ __bf16 Al[128][72];   // 128 rows x 64 k (+8 pad)
    __shared__ __bf16 Btl[64][72];   // 64 cols (n) x 64 k (+8 pad)

    const int t    = threadIdx.x;
    const int lane = t & 31;
    const int wave = t >> 5;
    const int r16  = lane & 15;
    const int hi   = lane >> 4;
    const int n0   = blockIdx.x * 64;
    const int row0 = blockIdx.y * 128;
    const int m0   = wave * 16;

    v8f acc[4] = {};

    for (int k0 = 0; k0 < NK; k0 += 64) {
        // ---- stage A tile (128 x 64) ----
        if (IN_BF16) {
            if (wave == 0) {
                tdm_load_2d_bf16(lds_off32(&Al[0][0]),
                                 (const __bf16*)Xv + (size_t)row0 * NK + k0,
                                 /*tensor*/64, 128, /*tile*/64, 128,
                                 /*stride*/NK, /*pad*/true);
            }
        } else {
            const float* X = (const float*)Xv;
            for (int idx = t; idx < 128 * 64; idx += 256) {
                int r = idx >> 6, c = idx & 63;
                Al[r][c] = (__bf16)X[(size_t)(row0 + r) * NK + k0 + c];
            }
        }
        // ---- stage B^T tile (64 n x 64 k), fp32 -> bf16, transpose ----
        for (int idx = t; idx < 64 * 64; idx += 256) {
            int r = idx >> 6, c = idx & 63;   // r = k, c = n
            Btl[c][r] = (__bf16)W[(size_t)(k0 + r) * NN + n0 + c];
        }
        // prefetch next weight tile (gfx1250 global_prefetch_b8)
        if (k0 + 64 < NK)
            __builtin_prefetch(&W[(size_t)(k0 + 64 + (t >> 2)) * NN + n0 + ((t & 3) << 4)], 0, 1);
        if (IN_BF16 && wave == 0)
            __builtin_amdgcn_s_wait_tensorcnt(0);
        __syncthreads();

        for (int kc = 0; kc < 2; ++kc) {
            // A fragment 16x32 (ISA layout: K = hi*8..+7 and 16+hi*8..+7)
            v8bf alo = *(const v8bf*)&Al[m0 + r16][kc * 32 + hi * 8];
            v8bf ahi = *(const v8bf*)&Al[m0 + r16][kc * 32 + 16 + hi * 8];
            v16bf a = cat16(alo, ahi);
            for (int nc = 0; nc < 4; ++nc) {
                // B fragment: lane = col n, K = hi*8..+7 and 16+hi*8..+7
                v8bf blo = *(const v8bf*)&Btl[nc * 16 + r16][kc * 32 + hi * 8];
                v8bf bhi = *(const v8bf*)&Btl[nc * 16 + r16][kc * 32 + 16 + hi * 8];
                acc[nc] = wmma_bf16(a, cat16(blo, bhi), acc[nc]);
            }
        }
        __syncthreads();
    }

    // ---- epilogue: C layout -> lane r16 = col, VGPR j = row j+8*hi ----
    for (int nc = 0; nc < 4; ++nc) {
        for (int j = 0; j < 8; ++j) {
            int m   = j + 8 * hi;
            int rr  = row0 + m0 + m;
            int col = n0 + nc * 16 + r16;
            float v = (acc[nc][j] + bias[col]) * scale;
            if (OUT_MODE == 0) {
                ((__bf16*)Out)[(size_t)rr * NN + col] = (__bf16)v;
            } else if (OUT_MODE == 1) {
                int b = rr >> 11, s = rr & (Ss - 1);
                int h = col >> 6, e = col & 63;
                ((__bf16*)Out)[((size_t)((b * Hh + h) * Hd + e)) * Ss + s] = (__bf16)v;
            } else {
                ((float*)Out)[(size_t)rr * NN + col] = v;
            }
        }
    }
}

// ---------------------------------------------------------------------------
// Attention: per (b, h, 128-row q tile). Two-pass exact softmax:
//   pass 1: QK^T via WMMA, online per-row max/sum in registers (no spill),
//           software-pipelined: next tile's WMMAs issue before the current
//           tile's max/sum VALU so hazard slots hold real work, not v_nop.
//   pass 2: recompute scores (pipelined), write normalized probs to d_out
//           once, accumulate O = P*V via WMMA (P staged through per-wave LDS
//           to convert C-fragment layout -> A-fragment layout).
// Q pre-scaled by 1/sqrt(Hd); Q tile staged by TDM.
// ---------------------------------------------------------------------------
__global__ __launch_bounds__(256)
void attn_kernel(const __bf16* __restrict__ Qbf, const __bf16* __restrict__ Kbf,
                 const __bf16* __restrict__ Vt, float* __restrict__ attnW,
                 __bf16* __restrict__ Xws)
{
    __shared__ __bf16 Ql[128][72];       // 128 q rows x 64 e (+8 pad, TDM-padded)
    __shared__ __bf16 Pl[8][16][32];     // per-wave P tile (16 q x 32 k)

    const int t    = threadIdx.x;
    const int lane = t & 31;
    const int wave = t >> 5;
    const int r16  = lane & 15;
    const int hi   = lane >> 4;
    const int q0   = blockIdx.x * 128;
    const int h    = blockIdx.y;
    const int b    = blockIdx.z;
    const int m0   = wave * 16;

    // ---- stage Q tile into LDS via Tensor Data Mover ----
    if (wave == 0) {
        tdm_load_2d_bf16(lds_off32(&Ql[0][0]),
                         Qbf + (size_t)(b * Ss + q0) * 1024 + h * 64,
                         /*tensor*/64, 128, /*tile*/64, 128,
                         /*stride*/1024, /*pad*/true);
        __builtin_amdgcn_s_wait_tensorcnt(0);
    }
    __syncthreads();

    // ---- Q A-fragments (2 chunks of head dim 64) held in registers ----
    v16bf aq[2];
    for (int c = 0; c < 2; ++c) {
        v8bf lo = *(const v8bf*)&Ql[m0 + r16][c * 32 + hi * 16];
        v8bf hh = *(const v8bf*)&Ql[m0 + r16][c * 32 + hi * 16 + 8];
        aq[c] = cat16(lo, hh);
    }

    const __bf16* Kbase = Kbf + (size_t)b * Ss * 1024 + h * 64;
    const __bf16* Vbase = Vt + ((size_t)(b * Hh + h) * Hd) * Ss;
    float* Arow = attnW + ((size_t)(b * Hh + h) * Ss) * Ss;

    // one 16-column score tile: S[16q x 16k] = Q(16x64) . K^T
    auto score16 = [&](int kc) -> v8f {
        v16bf bk0 = *(const v16bf*)(Kbase + (size_t)(kc + r16) * 1024 + hi * 16);
        v16bf bk1 = *(const v16bf*)(Kbase + (size_t)(kc + r16) * 1024 + 32 + hi * 16);
        v8f cs = {};
        cs = wmma_bf16(aq[0], bk0, cs);
        cs = wmma_bf16(aq[1], bk1, cs);
        return cs;
    };

    // ---- pass 1: per-lane online max/sum, software-pipelined ----
    float mx[8], lsum[8];
    for (int j = 0; j < 8; ++j) { mx[j] = -3.0e38f; lsum[j] = 0.0f; }

    v8f cs = score16(0);
    for (int k0 = 0; k0 < Ss; k0 += 16) {
        int kn = (k0 + 16 < Ss) ? (k0 + 16) : 0;   // last iter: harmless recompute
        v8f nx = score16(kn);                       // independent WMMAs fill hazard
        for (int j = 0; j < 8; ++j) {
            float v  = cs[j];
            float nm = fmaxf(mx[j], v);
            lsum[j]  = lsum[j] * __expf(mx[j] - nm) + __expf(v - nm);
            mx[j]    = nm;
        }
        cs = nx;
    }
    // cross-lane reduce within each 16-lane half (rows j+8*hi live per half)
    for (int off = 1; off < 16; off <<= 1) {
        for (int j = 0; j < 8; ++j) {
            float om = __shfl_xor(mx[j], off);
            float ol = __shfl_xor(lsum[j], off);
            float nm = fmaxf(mx[j], om);
            lsum[j]  = lsum[j] * __expf(mx[j] - nm) + ol * __expf(om - nm);
            mx[j]    = nm;
        }
    }
    float inv[8];
    for (int j = 0; j < 8; ++j) inv[j] = 1.0f / lsum[j];

    // ---- pass 2: recompute scores (pipelined), emit probs, O += P*V ----
    v8f o[4] = {};
    v8f cs0 = score16(0);
    v8f cs1 = score16(16);
    for (int k0 = 0; k0 < Ss; k0 += 32) {
        // transform + emit the two current 16-col half tiles
        for (int h2 = 0; h2 < 2; ++h2) {
            v8f c2 = h2 ? cs1 : cs0;
            int kc = k0 + 16 * h2;
            for (int j = 0; j < 8; ++j) {
                int m = j + 8 * hi;
                float p = __expf(c2[j] - mx[j]) * inv[j];
                Arow[(size_t)(q0 + m0 + m) * Ss + kc + r16] = p;   // attn weights out
                Pl[wave][m][16 * h2 + r16] = (__bf16)p;            // stage for P*V
            }
        }
        // issue next step's score WMMAs before consuming P (fills PV latency)
        int kn = (k0 + 32 < Ss) ? (k0 + 32) : 0;
        v8f ns0 = score16(kn);
        v8f ns1 = score16(kn + 16);
        // per-wave LDS store->load: DS ops from the same wave stay in order
        v8bf plo = *(const v8bf*)&Pl[wave][r16][hi * 16];
        v8bf phi = *(const v8bf*)&Pl[wave][r16][hi * 16 + 8];
        v16bf ap = cat16(plo, phi);
        for (int et = 0; et < 4; ++et) {
            v16bf bv = *(const v16bf*)(Vbase + (size_t)(et * 16 + r16) * Ss + k0 + hi * 16);
            o[et] = wmma_bf16(ap, bv, o[et]);
        }
        cs0 = ns0;
        cs1 = ns1;
    }

    // ---- store O as bf16 [bs][h*64+e] for the output projection ----
    for (int et = 0; et < 4; ++et) {
        for (int j = 0; j < 8; ++j) {
            int m = j + 8 * hi;
            Xws[(size_t)(b * Ss + q0 + m0 + m) * 1024 + h * 64 + et * 16 + r16] =
                (__bf16)(o[et][j]);
        }
    }
}

// ---------------------------------------------------------------------------
extern "C" void kernel_launch(void* const* d_in, const int* in_sizes, int n_in,
                              void* d_out, int out_size, void* d_ws, size_t ws_size,
                              hipStream_t stream)
{
    const float* inputs_q  = (const float*)d_in[0];
    const float* inputs_kv = (const float*)d_in[1];
    const float* Wq = (const float*)d_in[2];
    const float* bq = (const float*)d_in[3];
    const float* Wk = (const float*)d_in[4];
    const float* bk = (const float*)d_in[5];
    const float* Wv = (const float*)d_in[6];
    const float* bv = (const float*)d_in[7];
    const float* Wo = (const float*)d_in[8];
    const float* bo = (const float*)d_in[9];

    float* out   = (float*)d_out;                    // [B,S,D]
    float* attnW = out + (size_t)Bb * Ss * Dd;       // [B,H,S,S]

    char* ws = (char*)d_ws;
    const size_t SEG = (size_t)BS * 1024 * sizeof(__bf16);   // 8 MB
    __bf16* Qbf = (__bf16*)(ws);
    __bf16* Kbf = (__bf16*)(ws + SEG);
    __bf16* Vt  = (__bf16*)(ws + 2 * SEG);
    __bf16* Xws = (__bf16*)(ws + 3 * SEG);

    dim3 blk(256);
    dim3 ggrid(NN / 64, BS / 128);

    // QKV projections (Q pre-scaled by 1/sqrt(64) = 0.125)
    gemm_kernel<0, 0><<<ggrid, blk, 0, stream>>>(inputs_q,  Wq, bq, Qbf, 0.125f);
    gemm_kernel<0, 0><<<ggrid, blk, 0, stream>>>(inputs_kv, Wk, bk, Kbf, 1.0f);
    gemm_kernel<0, 1><<<ggrid, blk, 0, stream>>>(inputs_kv, Wv, bv, Vt,  1.0f);

    // attention + attention-weight materialization
    dim3 agrid(Ss / 128, Hh, Bb);
    attn_kernel<<<agrid, blk, 0, stream>>>(Qbf, Kbf, Vt, attnW, Xws);

    // output projection (fp32 result)
    gemm_kernel<1, 2><<<ggrid, blk, 0, stream>>>(Xws, Wo, bo, out, 1.0f);
}
